// TransformerEncoderLayer_45775761440800
// MI455X (gfx1250) — matrix operand order repaired and checked by reading, not compiled, not run on hardware
//
#include <hip/hip_runtime.h>
#include <hip/hip_bf16.h>
#include <math.h>
#include <stdint.h>

typedef __attribute__((ext_vector_type(16))) __bf16 v16bf;
typedef __attribute__((ext_vector_type(8)))  float  v8f;
typedef __attribute__((ext_vector_type(4)))  float  v4f;
typedef __attribute__((ext_vector_type(4)))  unsigned int u32x4;

#define N_TOK   16384
#define D_MODEL 256
#define D_FF    2048
#define NHEAD   8
#define HDIM    32
#define LNBR    32

#if __has_builtin(__builtin_amdgcn_s_wait_asynccnt)
#define WAIT_ASYNC(n) __builtin_amdgcn_s_wait_asynccnt(n)
#else
#define WAIT_ASYNC(n) asm volatile("s_wait_asynccnt %0" :: "i"(n) : "memory")
#endif

// ---------------------------------------------------------------------------
// One-time fp32 -> bf16 elementwise convert (vectorized by 4)
// ---------------------------------------------------------------------------
__global__ __launch_bounds__(256)
void f32_to_bf16_kernel(const float* __restrict__ in, __bf16* __restrict__ out)
{
    const size_t i = ((size_t)blockIdx.x * 256 + threadIdx.x) * 4;
    const v4f v = *reinterpret_cast<const v4f*>(in + i);
    out[i + 0] = (__bf16)v.x;
    out[i + 1] = (__bf16)v.y;
    out[i + 2] = (__bf16)v.z;
    out[i + 3] = (__bf16)v.w;
}

// ---------------------------------------------------------------------------
// One-time weight transpose + convert: in[K][Nc] fp32 -> out[Nc][K] bf16.
// 32x32 LDS tile, block (32,8).
// ---------------------------------------------------------------------------
__global__ __launch_bounds__(256)
void transpose_f32_to_bf16(const float* __restrict__ in, __bf16* __restrict__ out,
                           int K, int Nc)
{
    __shared__ float tile[32][33];
    const int c0 = blockIdx.x * 32;
    const int k0 = blockIdx.y * 32;
    #pragma unroll
    for (int i = 0; i < 4; ++i)
        tile[threadIdx.y + i * 8][threadIdx.x] =
            in[(size_t)(k0 + threadIdx.y + i * 8) * Nc + (c0 + threadIdx.x)];
    __syncthreads();
    #pragma unroll
    for (int i = 0; i < 4; ++i)
        out[(size_t)(c0 + threadIdx.y + i * 8) * K + (k0 + threadIdx.x)] =
            (__bf16)tile[threadIdx.x][threadIdx.y + i * 8];
}

// ---------------------------------------------------------------------------
// Tiled GEMM:  C[M,Nc] = act( A[M,K] @ W[K,Nc] + bias ) * postScale
// A: bf16 row-major [M,K].  WT: bf16 TRANSPOSED weights [Nc,K].
// Block tile 128x128, 8 waves (4 rows x 2 cols), each wave 32x64 = 2x4 WMMA
// frags, K-step 32. LDS double-buffered; tiles staged with
// GLOBAL_LOAD_ASYNC_TO_LDS_B128 (ASYNCcnt pipelined: 4 async ops per tile).
// ---------------------------------------------------------------------------
__global__ __launch_bounds__(256)
void gemm_bf16_wmma(const __bf16* __restrict__ A, const __bf16* __restrict__ WT,
                    const float* __restrict__ bias,
                    float* __restrict__ Cf, __bf16* __restrict__ Cbf, int c_is_bf16,
                    int M, int K, int Nc, float postScale, int relu)
{
    // row stride 40 elems (80 B): keeps every 8-elem chunk 16B-aligned, kills
    // bank conflicts between the two lane halves.
    __shared__ __bf16 As[2][128 * 40];
    __shared__ __bf16 Bs[2][128 * 40];

    const int tid  = threadIdx.x;
    const int lane = tid & 31;
    const int w    = tid >> 5;
    const int wr   = w & 3;           // wave row 0..3  -> rows wr*32
    const int wc   = w >> 2;          // wave col 0..1  -> cols wc*64
    const int rowBase = blockIdx.y * 128;
    const int colBase = blockIdx.x * 128;

    v8f acc[2][4];
    #pragma unroll
    for (int i = 0; i < 2; ++i)
        #pragma unroll
        for (int j = 0; j < 4; ++j)
            #pragma unroll
            for (int e = 0; e < 8; ++e) acc[i][j][e] = 0.0f;

    // stage one 128-row x 32-K bf16 tile (8 KB) via async copy:
    // 512 x 16B segments, 2 async instructions per thread.
    auto stage = [&](const __bf16* g, int gRowBase, int stride, int kb, __bf16* lds) {
        #pragma unroll
        for (int it = 0; it < 2; ++it) {
            const int s   = tid + it * 256;        // 0..511
            const int r   = s >> 2;                // 0..127
            const int seg = s & 3;                 // 16B segment in row
            const __bf16* gp = g + (size_t)(gRowBase + r) * (size_t)stride
                                 + (size_t)(kb + seg * 8);
            const unsigned lo = (unsigned)(uintptr_t)(lds + r * 40 + seg * 8);
            asm volatile("global_load_async_to_lds_b128 %0, %1, off"
                         :: "v"(lo), "v"(gp) : "memory");
        }
    };

    stage(A,  rowBase, K, 0, As[0]);
    stage(WT, colBase, K, 0, Bs[0]);

    const int half = lane >> 4;   // 0: lanes 0-15, 1: lanes 16-31
    const int l15  = lane & 15;
    const int nT   = K >> 5;

    for (int t = 0; t < nT; ++t) {
        const int cur = t & 1;
        if (t + 1 < nT) {
            stage(A,  rowBase, K, (t + 1) * 32, As[cur ^ 1]);
            stage(WT, colBase, K, (t + 1) * 32, Bs[cur ^ 1]);
            WAIT_ASYNC(4);        // in-order: previous tile's 4 ops complete
        } else {
            WAIT_ASYNC(0);
        }
        __syncthreads();

        // A frag (16-bit A 16x32 ISA layout): lanes0-15 elems0..7=K0..7,
        // 8..15=K16..23; lanes16-31 shifted by 8; row = lane&15.
        v16bf afr[2], bfr[4];
        #pragma unroll
        for (int i = 0; i < 2; ++i) {
            const __bf16* base = &As[cur][(wr * 32 + i * 16 + l15) * 40 + half * 8];
            union { v16bf v; u32x4 q[2]; } u;
            u.q[0] = *reinterpret_cast<const u32x4*>(base);
            u.q[1] = *reinterpret_cast<const u32x4*>(base + 16);
            afr[i] = u.v;
        }
        // B frag (32x16): col = lane&15; same K layout as A (lanes0-15 K0..7,16..23; lanes16-31 shifted by 8).
        #pragma unroll
        for (int j = 0; j < 4; ++j) {
            const __bf16* base = &Bs[cur][(wc * 64 + j * 16 + l15) * 40 + half * 8];
            union { v16bf v; u32x4 q[2]; } u;
            u.q[0] = *reinterpret_cast<const u32x4*>(base);
            u.q[1] = *reinterpret_cast<const u32x4*>(base + 16);
            bfr[j] = u.v;
        }

        #pragma unroll
        for (int i = 0; i < 2; ++i)
            #pragma unroll
            for (int j = 0; j < 4; ++j)
                acc[i][j] = __builtin_amdgcn_wmma_f32_16x16x32_bf16(
                    false, afr[i], false, bfr[j], (short)0, acc[i][j], false, false);

        __syncthreads();
    }

    // epilogue: C/D layout elem e -> M = e + 8*half, N = lane&15.
    #pragma unroll
    for (int i = 0; i < 2; ++i) {
        #pragma unroll
        for (int j = 0; j < 4; ++j) {
            const int col = colBase + wc * 64 + j * 16 + l15;
            const float bv = bias ? bias[col] : 0.0f;
            #pragma unroll
            for (int e = 0; e < 8; ++e) {
                const int row = rowBase + wr * 32 + i * 16 + e + 8 * half;
                float val = acc[i][j][e] + bv;
                if (relu) val = fmaxf(val, 0.0f);
                val *= postScale;
                const size_t o = (size_t)row * (size_t)Nc + (size_t)col;
                if (c_is_bf16) Cbf[o] = (__bf16)val;
                else           Cf[o]  = val;
            }
        }
    }
}

// ---------------------------------------------------------------------------
// Gather attention: one wave per query, 32 neighbors, 8 heads of dim 32.
// fp32 math; bf16 output (feeds the Wo GEMM).
// ---------------------------------------------------------------------------
__global__ __launch_bounds__(256)
void attn_gather_kernel(const float* __restrict__ q, const float* __restrict__ k,
                        const float* __restrict__ v,
                        const int* __restrict__ index_pair,
                        const int* __restrict__ index_pair_batch,
                        const int* __restrict__ key_batch_cnt,
                        __bf16* __restrict__ out)
{
    __shared__ float qs[8][D_MODEL];
    __shared__ float attn_s[8][NHEAD * LNBR];
    __shared__ int   rows_s[8][LNBR];

    const int w    = threadIdx.x >> 5;
    const int lane = threadIdx.x & 31;
    const int n    = blockIdx.x * 8 + w;

    const int bidx = index_pair_batch[n];
    int off = 0;
    for (int b = 0; b < bidx; ++b) off += key_batch_cnt[b];

    #pragma unroll
    for (int i = 0; i < 8; ++i)
        qs[w][i * 32 + lane] = q[(size_t)n * D_MODEL + i * 32 + lane];

    const int  idx   = index_pair[n * LNBR + lane];
    const bool valid = (idx >= 0);
    const int  row   = valid ? (idx + off) : 0;
    rows_s[w][lane]  = row;
    __syncthreads();

    float lg[NHEAD];
    const float* krow = k + (size_t)row * D_MODEL;
    #pragma unroll
    for (int h = 0; h < NHEAD; ++h) {
        float s = 0.0f;
        #pragma unroll
        for (int d = 0; d < HDIM; ++d)
            s += qs[w][h * HDIM + d] * krow[h * HDIM + d];
        lg[h] = valid ? s : -1e9f;
    }

    #pragma unroll
    for (int h = 0; h < NHEAD; ++h) {
        float m = lg[h];
        for (int o = 16; o > 0; o >>= 1) m = fmaxf(m, __shfl_xor(m, o, 32));
        float e = __expf(lg[h] - m);
        float s = e;
        for (int o = 16; o > 0; o >>= 1) s += __shfl_xor(s, o, 32);
        attn_s[w][h * LNBR + lane] = e / s;
    }
    __syncthreads();

    #pragma unroll
    for (int h = 0; h < NHEAD; ++h) {
        float acc = 0.0f;
        #pragma unroll 4
        for (int j = 0; j < LNBR; ++j) {
            const int r = rows_s[w][j];
            acc += attn_s[w][h * LNBR + j] * v[(size_t)r * D_MODEL + h * HDIM + lane];
        }
        out[(size_t)n * D_MODEL + h * HDIM + lane] = (__bf16)acc;
    }
}

// ---------------------------------------------------------------------------
// Fused residual add + LayerNorm (rows of 256). One wave per row.
// Optional bf16 mirror of the output (feeds next GEMM).
// ---------------------------------------------------------------------------
__global__ __launch_bounds__(256)
void add_layernorm_kernel(const float* __restrict__ a, const float* __restrict__ b,
                          const float* __restrict__ g, const float* __restrict__ beta,
                          float* __restrict__ out, __bf16* __restrict__ out_bf)
{
    const int w    = threadIdx.x >> 5;
    const int lane = threadIdx.x & 31;
    const int n    = blockIdx.x * 8 + w;

    float x[8];
    float s = 0.0f, ss = 0.0f;
    #pragma unroll
    for (int i = 0; i < 8; ++i) {
        const int c = i * 32 + lane;
        const float t = a[(size_t)n * D_MODEL + c] + b[(size_t)n * D_MODEL + c];
        x[i] = t;
        s  += t;
        ss += t * t;
    }
    for (int o = 16; o > 0; o >>= 1) {
        s  += __shfl_xor(s,  o, 32);
        ss += __shfl_xor(ss, o, 32);
    }
    const float mean = s * (1.0f / D_MODEL);
    const float var  = ss * (1.0f / D_MODEL) - mean * mean;
    const float rstd = rsqrtf(var + 1e-5f);
    #pragma unroll
    for (int i = 0; i < 8; ++i) {
        const int c = i * 32 + lane;
        const float y = (x[i] - mean) * rstd * g[c] + beta[c];
        out[(size_t)n * D_MODEL + c] = y;
        if (out_bf) out_bf[(size_t)n * D_MODEL + c] = (__bf16)y;
    }
}

// ---------------------------------------------------------------------------
extern "C" void kernel_launch(void* const* d_in, const int* in_sizes, int n_in,
                              void* d_out, int out_size, void* d_ws, size_t ws_size,
                              hipStream_t stream)
{
    (void)in_sizes; (void)n_in; (void)out_size; (void)ws_size;

    const float* src = (const float*)d_in[0];
    const int*   index_pair       = (const int*)d_in[1];
    const int*   key_batch_cnt    = (const int*)d_in[3];
    const int*   index_pair_batch = (const int*)d_in[4];
    const float* Wq = (const float*)d_in[5];  const float* bq = (const float*)d_in[6];
    const float* Wk = (const float*)d_in[7];  const float* bk = (const float*)d_in[8];
    const float* Wv = (const float*)d_in[9];  const float* bv = (const float*)d_in[10];
    const float* Wo = (const float*)d_in[11]; const float* bo = (const float*)d_in[12];
    const float* W1 = (const float*)d_in[13]; const float* b1 = (const float*)d_in[14];
    const float* W2 = (const float*)d_in[15]; const float* b2 = (const float*)d_in[16];
    const float* g1 = (const float*)d_in[17]; const float* be1 = (const float*)d_in[18];
    const float* g2 = (const float*)d_in[19]; const float* be2 = (const float*)d_in[20];

    const size_t ND = (size_t)N_TOK * D_MODEL;      // 4 Mi elems
    const size_t NF = (size_t)N_TOK * D_FF;         // 32 Mi elems

    // ---- workspace carve (256 B aligned) ----
    char* cur = (char*)d_ws;
    auto carve = [&](size_t bytes) -> void* {
        void* p = cur;
        cur += (bytes + 255) & ~(size_t)255;
        return p;
    };
    __bf16* src_bf = (__bf16*)carve(ND * 2);
    __bf16* WTq    = (__bf16*)carve((size_t)D_MODEL * D_MODEL * 2);
    __bf16* WTk    = (__bf16*)carve((size_t)D_MODEL * D_MODEL * 2);
    __bf16* WTv    = (__bf16*)carve((size_t)D_MODEL * D_MODEL * 2);
    __bf16* WTo    = (__bf16*)carve((size_t)D_MODEL * D_MODEL * 2);
    __bf16* WT1    = (__bf16*)carve((size_t)D_MODEL * D_FF * 2);
    __bf16* WT2    = (__bf16*)carve((size_t)D_FF * D_MODEL * 2);
    float*  q      = (float*) carve(ND * 4);
    float*  k      = (float*) carve(ND * 4);
    float*  v      = (float*) carve(ND * 4);
    __bf16* attnb  = (__bf16*)carve(ND * 2);
    float*  src2   = (float*) carve(ND * 4);
    float*  x      = (float*) carve(ND * 4);
    __bf16* x_bf   = (__bf16*)carve(ND * 2);
    __bf16* h      = (__bf16*)carve(NF * 2);
    float*  ff     = (float*) carve(ND * 4);

    const float qscale = 0.17677669529663687f;      // 1/sqrt(32)

    const dim3 blk(256);
    const dim3 tblk(32, 8);

    // one-time precision prep: src -> bf16, weights -> transposed bf16
    f32_to_bf16_kernel<<<ND / 1024, blk, 0, stream>>>(src, src_bf);
    transpose_f32_to_bf16<<<dim3(D_MODEL/32, D_MODEL/32), tblk, 0, stream>>>(Wq, WTq, D_MODEL, D_MODEL);
    transpose_f32_to_bf16<<<dim3(D_MODEL/32, D_MODEL/32), tblk, 0, stream>>>(Wk, WTk, D_MODEL, D_MODEL);
    transpose_f32_to_bf16<<<dim3(D_MODEL/32, D_MODEL/32), tblk, 0, stream>>>(Wv, WTv, D_MODEL, D_MODEL);
    transpose_f32_to_bf16<<<dim3(D_MODEL/32, D_MODEL/32), tblk, 0, stream>>>(Wo, WTo, D_MODEL, D_MODEL);
    transpose_f32_to_bf16<<<dim3(D_FF/32,    D_MODEL/32), tblk, 0, stream>>>(W1, WT1, D_MODEL, D_FF);
    transpose_f32_to_bf16<<<dim3(D_MODEL/32, D_FF/32),    tblk, 0, stream>>>(W2, WT2, D_FF, D_MODEL);

    const dim3 gD(D_MODEL / 128, N_TOK / 128);      // Nc=256 GEMMs
    const dim3 gF(D_FF   / 128, N_TOK / 128);       // Nc=2048 GEMM

    // Q, K, V projections (q gets 1/sqrt(hd) folded into the epilogue)
    gemm_bf16_wmma<<<gD, blk, 0, stream>>>(src_bf, WTq, bq, q, nullptr, 0,
                                           N_TOK, D_MODEL, D_MODEL, qscale, 0);
    gemm_bf16_wmma<<<gD, blk, 0, stream>>>(src_bf, WTk, bk, k, nullptr, 0,
                                           N_TOK, D_MODEL, D_MODEL, 1.0f, 0);
    gemm_bf16_wmma<<<gD, blk, 0, stream>>>(src_bf, WTv, bv, v, nullptr, 0,
                                           N_TOK, D_MODEL, D_MODEL, 1.0f, 0);

    // gathered local attention (fp32 softmax, bf16 out)
    attn_gather_kernel<<<N_TOK / 8, blk, 0, stream>>>(q, k, v, index_pair,
                                                      index_pair_batch, key_batch_cnt,
                                                      attnb);

    // output projection, residual + LN1 (also emits bf16 x for FFN GEMM)
    gemm_bf16_wmma<<<gD, blk, 0, stream>>>(attnb, WTo, bo, src2, nullptr, 0,
                                           N_TOK, D_MODEL, D_MODEL, 1.0f, 0);
    add_layernorm_kernel<<<N_TOK / 8, blk, 0, stream>>>(src, src2, g1, be1, x, x_bf);

    // FFN: relu(x@W1+b1) kept bf16, then @W2+b2, residual + LN2 -> d_out
    gemm_bf16_wmma<<<gF, blk, 0, stream>>>(x_bf, WT1, b1, nullptr, h, 1,
                                           N_TOK, D_MODEL, D_FF, 1.0f, 1);
    gemm_bf16_wmma<<<gD, blk, 0, stream>>>(h, WT2, b2, ff, nullptr, 0,
                                           N_TOK, D_FF, D_MODEL, 1.0f, 0);
    add_layernorm_kernel<<<N_TOK / 8, blk, 0, stream>>>(x, ff, g2, be2,
                                                        (float*)d_out, nullptr);
}
